// DSConv_55637006352785
// MI455X (gfx1250) — matrix operand order repaired and checked by reading, not compiled, not run on hardware
//
#include <hip/hip_runtime.h>
#include <hip/hip_bf16.h>
#include <math.h>

// Problem constants (match reference)
#define Bn   4
#define Cc   64
#define Wd   128
#define Hd   128
#define Kk   9
#define OCn  64
#define KKtot 576           // Kk * Cc  (GEMM reduction length)
#define SK    584           // padded LDS row stride in bf16 elems (conflict-free)

typedef __attribute__((ext_vector_type(16))) __bf16 v16bf;
typedef __attribute__((ext_vector_type(8)))  __bf16 bf8;
typedef __attribute__((ext_vector_type(8)))  float  v8f;

// gfx1250 async Global->LDS copy (ASYNCcnt-tracked), guarded so compile never
// regresses if the builtin is absent. Param type per hipcc diagnostic:
// int __vector_size__(16) in addrspace(1) (src) / addrspace(3) (dst).
#if __has_builtin(__builtin_amdgcn_global_load_async_to_lds_b128) && \
    __has_builtin(__builtin_amdgcn_s_wait_asynccnt)
#define USE_ASYNC_LDS 1
typedef __attribute__((vector_size(16))) int i32x4;
typedef __attribute__((address_space(1))) i32x4 g_i32x4;
typedef __attribute__((address_space(3))) i32x4 l_i32x4;
#else
#define USE_ASYNC_LDS 0
#endif

// ---------------------------------------------------------------------------
// Kernel 0a: f (B,C,W,H) f32  ->  fn (B,W,H,C) bf16   (channels contiguous)
// ---------------------------------------------------------------------------
__global__ void k_transpose(const float* __restrict__ f, __bf16* __restrict__ fn) {
  __shared__ float t[Hd][Cc + 1];
  int blk = blockIdx.x;              // b*Wd + wy
  int b = blk / Wd, wy = blk % Wd;
  for (int idx = threadIdx.x; idx < Cc * Hd; idx += blockDim.x) {
    int c = idx / Hd, hx = idx % Hd;                 // coalesced in hx
    t[hx][c] = f[((b * Cc + c) * Wd + wy) * Hd + hx];
  }
  __syncthreads();
  for (int idx = threadIdx.x; idx < Hd * Cc; idx += blockDim.x) {
    int hx = idx / Cc, c = idx % Cc;                 // coalesced in c
    fn[((b * Wd + wy) * Hd + hx) * Cc + c] = (__bf16)t[hx][c];
  }
}

// ---------------------------------------------------------------------------
// Kernel 0b: conv_w (OC,C,K,1) f32 -> W_pack[oc][tap*64+c] bf16
// ---------------------------------------------------------------------------
__global__ void k_packw(const float* __restrict__ conv_w, __bf16* __restrict__ wp) {
  int i = blockIdx.x * blockDim.x + threadIdx.x;
  if (i >= OCn * KKtot) return;
  int oc  = i / KKtot;
  int kk  = i % KKtot;
  int tap = kk >> 6;
  int c   = kk & 63;
  wp[i] = (__bf16)conv_w[(oc * Cc + c) * Kk + tap];
}

// ---------------------------------------------------------------------------
// Kernel 1: offset conv (3x3, C=64 -> 9) + BN + tanh + snake cumsum -> y_cum
// Full f32 precision (these become sampling coordinates).
// ---------------------------------------------------------------------------
__global__ void k_offset(const float* __restrict__ f,
                         const float* __restrict__ offset_w,
                         const float* __restrict__ offset_b,
                         const float* __restrict__ bn_gamma,
                         const float* __restrict__ bn_beta,
                         const float* __restrict__ bn_mean,
                         const float* __restrict__ bn_var,
                         float* __restrict__ y_cum) {
  __shared__ float wl[9 * 64 * 9];                   // [tap(kh*3+kw)][c][k]
  for (int i = threadIdx.x; i < 9 * 64 * 9; i += blockDim.x) {
    int k   = i % 9;
    int c   = (i / 9) % 64;
    int tap = i / (9 * 64);
    int kh = tap / 3, kw = tap % 3;
    wl[i] = offset_w[((k * Cc + c) * 3 + kh) * 3 + kw];
  }
  __syncthreads();
  int pix = blockIdx.x * blockDim.x + threadIdx.x;   // 65536 pixels
  int b  = pix >> 14;
  int wy = (pix >> 7) & 127;
  int hx = pix & 127;

  float acc[9];
  #pragma unroll
  for (int k = 0; k < 9; ++k) acc[k] = 0.f;

  for (int c = 0; c < Cc; ++c) {
    const float* fc = f + ((b * Cc + c) * Wd) * Hd;
    #pragma unroll
    for (int kh = 0; kh < 3; ++kh) {
      int y = wy + kh - 1;
      if ((unsigned)y >= (unsigned)Wd) continue;     // zero pad (SAME)
      #pragma unroll
      for (int kw = 0; kw < 3; ++kw) {
        int x = hx + kw - 1;
        if ((unsigned)x >= (unsigned)Hd) continue;
        float v = fc[y * Hd + x];
        const float* wp = &wl[((kh * 3 + kw) * 64 + c) * 9];
        #pragma unroll
        for (int k = 0; k < 9; ++k) acc[k] += v * wp[k];
      }
    }
  }

  float t[9];
  #pragma unroll
  for (int k = 0; k < 9; ++k) {
    float v = acc[k] + offset_b[k];
    v = (v - bn_mean[k]) * rsqrtf(bn_var[k] + 1e-5f);
    v = v * bn_gamma[k] + bn_beta[k];
    t[k] = tanhf(v);
  }
  // Reference cumsum quirk: new[0] and new[8] stay raw tanh outputs.
  float n[9];
  n[4] = 0.f;
  n[5] = t[5]; n[6] = n[5] + t[6]; n[7] = n[6] + t[7]; n[8] = t[8];
  n[3] = t[3]; n[2] = n[3] + t[2]; n[1] = n[2] + t[1]; n[0] = t[0];
  #pragma unroll
  for (int k = 0; k < 9; ++k)
    y_cum[((b * Kk + k) * Wd + wy) * Hd + hx] = n[k];
}

// ---------------------------------------------------------------------------
// Kernel 2: bilinear snake gather (phase 1, to LDS bf16) + WMMA GEMM (phase 2)
// Block = (b, w, 64 h-pixels); 128 threads (4 waves); output 64oc x 64px.
// D = A(64x576 weights) * B(576x64 gathered samples), via 16x16x32 bf16 WMMA.
// Weights stream Global->LDS asynchronously (ASYNCcnt) overlapped with gather.
// ---------------------------------------------------------------------------
__global__ void __launch_bounds__(128)
k_dsconv_wmma(const __bf16* __restrict__ fn,
              const float*  __restrict__ y_cum,
              const __bf16* __restrict__ wp,
              const float*  __restrict__ conv_b,
              float*        __restrict__ z) {
  extern __shared__ __bf16 lds[];
  __bf16* Wl = lds;                  // [64 oc rows][SK]
  __bf16* Bt = lds + 64 * SK;        // [64 px rows][SK]

  int tid = threadIdx.x;
  int blk = blockIdx.x;              // ((b*Wd + w)*2 + ht)
  int ht = blk & 1;
  int w  = (blk >> 1) & 127;
  int b  = blk >> 8;
  int h0 = ht * 64;

  // ---- stage weights into LDS (bf16, padded stride) ----
  for (int q = tid; q < 64 * 72; q += 128) {         // 72 chunks of 8 bf16 per row
    int row = q / 72, cj = q % 72;
#if USE_ASYNC_LDS
    __builtin_amdgcn_global_load_async_to_lds_b128(
        (g_i32x4*)(const_cast<__bf16*>(wp) + row * KKtot + cj * 8),
        (l_i32x4*)(Wl + row * SK + cj * 8), 0, 0);
#else
    *(bf8*)(Wl + row * SK + cj * 8) = *(const bf8*)(wp + row * KKtot + cj * 8);
#endif
  }

  // ---- phase 1: gather. 8-lane groups, one (px,tap) per group per iter ----
  int grp = tid >> 3;                // 0..15
  int lc  = tid & 7;                 // 8 channels per lane
  const float lin0 = -5.0f;          // linspace(-5, 4, 9) -> step 1.125
  for (int it = 0; it < 36; ++it) {
    int pair = it * 16 + grp;        // 0..575
    int px   = pair / 9;             // 0..63
    int tap  = pair - px * 9;        // 0..8
    int h = h0 + px;
    float ys = (float)w + y_cum[((b * Kk + tap) * Wd + w) * Hd + h];
    float xs = (float)h + (lin0 + 1.125f * (float)tap);
    int y0 = (int)floorf(ys); y0 = min(max(y0, 0), Wd - 1);
    int y1 = min(y0 + 1, Wd - 1);
    int x0 = (int)floorf(xs); x0 = min(max(x0, 0), Hd - 1);
    int x1 = min(x0 + 1, Hd - 1);
    float y0f = (float)y0, y1f = (float)y1, x0f = (float)x0, x1f = (float)x1;
    float wa0 = (y1f - ys) * (x1f - xs);
    float wa1 = (y1f - ys) * (xs - x0f);
    float wb0 = (ys - y0f) * (x1f - xs);
    float wb1 = (ys - y0f) * (xs - x0f);

    int co = lc << 3;
    bf8 a0 = *(const bf8*)(fn + (((b * Wd + y0) * Hd + x0) << 6) + co);
    bf8 a1 = *(const bf8*)(fn + (((b * Wd + y0) * Hd + x1) << 6) + co);
    bf8 c0 = *(const bf8*)(fn + (((b * Wd + y1) * Hd + x0) << 6) + co);
    bf8 c1 = *(const bf8*)(fn + (((b * Wd + y1) * Hd + x1) << 6) + co);
    bf8 o;
    #pragma unroll
    for (int i = 0; i < 8; ++i) {
      float r = wa0 * (float)a0[i] + wa1 * (float)a1[i]
              + wb0 * (float)c0[i] + wb1 * (float)c1[i];
      o[i] = (__bf16)r;
    }
    *(bf8*)(Bt + px * SK + tap * 64 + co) = o;       // kk = tap*64 + c
  }

#if USE_ASYNC_LDS
  __builtin_amdgcn_s_wait_asynccnt(0);               // weights landed in LDS
#endif
  __syncthreads();

  // ---- phase 2: WMMA. Each wave: 16 pixels, all 64 oc (4 M-tiles) ----
  int wave = tid >> 5;
  int lane = tid & 31;
  int mrow = lane & 15;              // B column (pixel) / A row (oc) / D column
  int half = lane >> 4;
  const __bf16* Bw = Bt + (wave * 16 + mrow) * SK;

  v8f acc[4] = {};
  for (int ks = 0; ks < 18; ++ks) {
    int kb = ks * 32;
    // B fragment: lane column n=mrow, elems j -> k = kb + half*16 + j
    const __bf16* bp = Bw + kb + half * 16;
    bf8 blo = *(const bf8*)bp;
    bf8 bhi = *(const bf8*)(bp + 8);
    v16bf bfrag = {};
    #pragma unroll
    for (int i = 0; i < 8; ++i) { bfrag[i] = blo[i]; bfrag[i + 8] = bhi[i]; }

    #pragma unroll
    for (int mt = 0; mt < 4; ++mt) {
      // A fragment: row m=mrow, elems j -> k = kb + half*16 + j
      const __bf16* ap = Wl + (mt * 16 + mrow) * SK + kb + half * 16;
      bf8 alo = *(const bf8*)ap;
      bf8 ahi = *(const bf8*)(ap + 8);
      v16bf afrag = {};
      #pragma unroll
      for (int i = 0; i < 8; ++i) { afrag[i] = alo[i]; afrag[i + 8] = ahi[i]; }

      acc[mt] = __builtin_amdgcn_wmma_f32_16x16x32_bf16(
          false, afrag, false, bfrag, (short)0, acc[mt], false, false);
    }
  }

  // ---- writeout: D layout -> lane col n=mrow is pixel; VGPR r -> M=r+8*half ----
  int hpx = h0 + wave * 16 + mrow;
  #pragma unroll
  for (int mt = 0; mt < 4; ++mt) {
    #pragma unroll
    for (int r = 0; r < 8; ++r) {
      int oc = mt * 16 + r + half * 8;
      z[((b * OCn + oc) * Wd + w) * Hd + hpx] = acc[mt][r] + conv_b[oc];
    }
  }
}

// ---------------------------------------------------------------------------
// Kernel 3a: per-(b,group) mean / rstd over 4 channels x W x H (deterministic)
// ---------------------------------------------------------------------------
__global__ void k_gnstats(const float* __restrict__ z, float* __restrict__ stats) {
  __shared__ float ssum[256], ssq[256];
  int bg = blockIdx.x;               // b*16 + g
  int b = bg >> 4, g = bg & 15;
  float s = 0.f, q = 0.f;
  const float* zp = z + ((size_t)(b * OCn + g * 4) * Wd * Hd);
  for (int idx = threadIdx.x; idx < 4 * Wd * Hd; idx += 256) {
    float v = zp[idx];
    s += v; q += v * v;
  }
  ssum[threadIdx.x] = s; ssq[threadIdx.x] = q;
  __syncthreads();
  for (int off = 128; off > 0; off >>= 1) {
    if ((int)threadIdx.x < off) {
      ssum[threadIdx.x] += ssum[threadIdx.x + off];
      ssq[threadIdx.x]  += ssq[threadIdx.x + off];
    }
    __syncthreads();
  }
  if (threadIdx.x == 0) {
    float inv  = 1.0f / (4.0f * Wd * Hd);
    float mean = ssum[0] * inv;
    float var  = ssq[0] * inv - mean * mean;
    stats[bg * 2 + 0] = mean;
    stats[bg * 2 + 1] = rsqrtf(var + 1e-5f);
  }
}

// ---------------------------------------------------------------------------
// Kernel 3b: normalize + gamma/beta + ReLU -> out
// ---------------------------------------------------------------------------
__global__ void k_gnapply(const float* __restrict__ z, const float* __restrict__ stats,
                          const float* __restrict__ gamma, const float* __restrict__ beta,
                          float* __restrict__ out) {
  int o = blockIdx.x * blockDim.x + threadIdx.x;
  if (o >= Bn * OCn * Wd * Hd) return;
  int oc = (o >> 14) & 63;
  int b  = o >> 20;
  int bg = b * 16 + (oc >> 2);
  float mean = stats[bg * 2], rstd = stats[bg * 2 + 1];
  float v = (z[o] - mean) * rstd * gamma[oc] + beta[oc];
  out[o] = fmaxf(v, 0.f);
}

// ---------------------------------------------------------------------------
extern "C" void kernel_launch(void* const* d_in, const int* in_sizes, int n_in,
                              void* d_out, int out_size, void* d_ws, size_t ws_size,
                              hipStream_t stream) {
  const float* f        = (const float*)d_in[0];
  const float* offset_w = (const float*)d_in[1];
  const float* offset_b = (const float*)d_in[2];
  const float* bn_gamma = (const float*)d_in[3];
  const float* bn_beta  = (const float*)d_in[4];
  const float* bn_mean  = (const float*)d_in[5];
  const float* bn_var   = (const float*)d_in[6];
  const float* conv_w   = (const float*)d_in[7];
  const float* conv_b   = (const float*)d_in[8];
  const float* gn_gamma = (const float*)d_in[9];
  const float* gn_beta  = (const float*)d_in[10];
  float* out = (float*)d_out;

  // Workspace layout (all 256B aligned):
  char* ws = (char*)d_ws;
  __bf16* fn    = (__bf16*)(ws);                       //  8,388,608 B
  float*  y_cum = (float*)(ws + 8388608);              //  2,359,296 B
  __bf16* wpack = (__bf16*)(ws + 10747904);            //     73,728 B
  float*  zbuf  = (float*)(ws + 10821632);             // 16,777,216 B
  float*  stats = (float*)(ws + 27598848);             //        512 B

  k_transpose<<<Bn * Wd, 256, 0, stream>>>(f, fn);
  k_packw<<<(OCn * KKtot) / 256, 256, 0, stream>>>(conv_w, wpack);
  k_offset<<<(Bn * Wd * Hd) / 256, 256, 0, stream>>>(
      f, offset_w, offset_b, bn_gamma, bn_beta, bn_mean, bn_var, y_cum);

  size_t smem = (size_t)(2 * 64 * SK) * sizeof(__bf16);   // 149,504 B
  k_dsconv_wmma<<<Bn * Wd * 2, 128, smem, stream>>>(fn, y_cum, wpack, conv_b, zbuf);

  k_gnstats<<<Bn * 16, 256, 0, stream>>>(zbuf, stats);
  k_gnapply<<<(Bn * OCn * Wd * Hd) / 256, 256, 0, stream>>>(
      zbuf, stats, gn_gamma, gn_beta, out);
}
